// NGPTMLACausalAttention_64424509440422
// MI455X (gfx1250) — compile-verified, matrix operand order checked by reading, not run on hardware
//
#include <hip/hip_runtime.h>

// ---------------- problem dims ----------------
#define D_    2048
#define NH_   16
#define DH_   128
#define DHR_  64
#define DC_   1024
#define B_    4
#define L_    2048
#define BL_   8192          // B*L
#define QKD_  192           // DH + DHR

typedef __bf16 bf16;
typedef __attribute__((ext_vector_type(16))) __bf16 v16bf;
typedef __attribute__((ext_vector_type(8)))  __bf16 bf16x8;
typedef __attribute__((ext_vector_type(8)))  float  v8f;

union FragU { v16bf v; bf16x8 h[2]; };

// Load a 16x32 bf16 WMMA operand fragment for this lane: two contiguous
// 8-element (16B) runs at p and p+16 (matches ISA 16-bit A/B layout where
// lane's K values are {kb..kb+7} and {kb+16..kb+23}).
__device__ __forceinline__ v16bf load_frag16(const bf16* p) {
  FragU u;
  u.h[0] = *(const bf16x8*)(p);
  u.h[1] = *(const bf16x8*)(p + 16);
  return u.v;
}

__device__ __forceinline__ v8f wmma_bf16f32(v16bf a, v16bf b, v8f c) {
  // (neg_a, A, neg_b, B, c_mod, C, reuse_a, reuse_b)
  return __builtin_amdgcn_wmma_f32_16x16x32_bf16(false, a, false, b, (short)0, c,
                                                 false, false);
}

// ---------------- fp32 -> bf16 cast ----------------
__global__ __launch_bounds__(256) void cast_f32_to_bf16(const float* __restrict__ in,
                                                        bf16* __restrict__ out, int n) {
  for (int i = blockIdx.x * blockDim.x + threadIdx.x; i < n;
       i += gridDim.x * blockDim.x)
    out[i] = (bf16)in[i];
}

// ---------------- tiled bf16 GEMM (C = A x B), fp32 accumulate ----------------
#define GBM 128
#define GBN 128
#define GBK 32

// Dynamic LDS only in this kernel: with no static __shared__, the dynamic
// segment starts at wave-LDS offset 0, so raw byte offsets can be handed to
// GLOBAL_LOAD_ASYNC_TO_LDS (its LDS operand is an offset from LDS base).
extern __shared__ char gemm_smem[];

template <typename OutT>
__global__ __launch_bounds__(256) void gemm_bf16_wmma(const bf16* __restrict__ A,
                                                      const bf16* __restrict__ B,
                                                      OutT* __restrict__ C,
                                                      int M, int N, int K) {
  bf16* As = (bf16*)gemm_smem;                                   // [GBM][GBK]
  bf16* Bs = (bf16*)(gemm_smem + (size_t)GBM * GBK * sizeof(bf16)); // [GBN][GBK] (transposed)

  const int tid  = threadIdx.x;
  const int lane = tid & 31;
  const int wave = tid >> 5;            // 0..7
  const int wm   = (wave >> 1) * 32;    // wave M offset: 0,32,64,96
  const int wn   = (wave & 1) * 64;     // wave N offset: 0,64
  const int bm   = blockIdx.y * GBM;
  const int bn   = blockIdx.x * GBN;

  const int hi = lane >> 4;             // half-wave select
  const int ln = lane & 15;
  const int kb = hi * 8;                // per-lane K base

  v8f acc[2][4];
#pragma unroll
  for (int i = 0; i < 2; ++i)
#pragma unroll
    for (int j = 0; j < 4; ++j) acc[i][j] = {};

  for (int k0 = 0; k0 < K; k0 += GBK) {
    __syncthreads();
    // A tile -> LDS via async DMA (ASYNCcnt path, no VGPR staging).
    // Chunk i covers As row chunk>>2, cols (chunk&3)*8.. -> LDS offset chunk*16.
#pragma unroll
    for (int i = 0; i < 2; ++i) {
      int chunk = tid + 256 * i;        // 512 chunks of 8 bf16 (16B)
      int r = chunk >> 2;
      int c = (chunk & 3) * 8;
      const bf16* gp = &A[(size_t)(bm + r) * K + k0 + c];
      unsigned ldsoff = (unsigned)(chunk * 16);
      asm volatile("global_load_async_to_lds_b128 %0, %1, off"
                   :: "v"(ldsoff), "v"(gp)
                   : "memory");
    }
    // B tile -> LDS transposed (gather; stays on the VGPR path)
#pragma unroll
    for (int i = 0; i < 2; ++i) {
      int task = tid + 256 * i;         // n in [0,128) x kg in {0,8,16,24}
      int n  = task & 127;
      int kg = (task >> 7) * 8;
      int gn = bn + n;
      bf16x8 t;
      if (gn < N) {
#pragma unroll
        for (int j = 0; j < 8; ++j) t[j] = B[(size_t)(k0 + kg + j) * N + gn];
        // prefetch next K-step's strided B lines into cache
        if (k0 + GBK < K)
          __builtin_prefetch(&B[(size_t)(k0 + GBK + kg) * N + gn], 0, 0);
      } else {
#pragma unroll
        for (int j = 0; j < 8; ++j) t[j] = (bf16)0.0f;
      }
      *(bf16x8*)&Bs[n * GBK + kg] = t;
    }
    asm volatile("s_wait_asynccnt 0x0" ::: "memory");  // async A-tile landed in LDS
    __syncthreads();

    v16bf af[2], bfr[4];
#pragma unroll
    for (int ms = 0; ms < 2; ++ms)
      af[ms]  = load_frag16(&As[(wm + ms * 16 + ln) * GBK + kb]);
#pragma unroll
    for (int ns = 0; ns < 4; ++ns)
      bfr[ns] = load_frag16(&Bs[(wn + ns * 16 + ln) * GBK + kb]);
#pragma unroll
    for (int ms = 0; ms < 2; ++ms)
#pragma unroll
      for (int ns = 0; ns < 4; ++ns)
        acc[ms][ns] = wmma_bf16f32(af[ms], bfr[ns], acc[ms][ns]);
  }

  // C/D layout: lane holds col = ln, rows r + 8*hi
#pragma unroll
  for (int ms = 0; ms < 2; ++ms)
#pragma unroll
    for (int ns = 0; ns < 4; ++ns)
#pragma unroll
      for (int r = 0; r < 8; ++r) {
        int row = bm + wm + ms * 16 + r + 8 * hi;
        int col = bn + wn + ns * 16 + ln;
        if (col < N) C[(size_t)row * N + col] = (OutT)acc[ms][ns][r];
      }
}

// ---------------- RoPE + concat + L2 normalize (+ fold s_qk into q) ----------------
// One 32-lane wave per (b,l,h); each lane owns 6 of the 192 channels.
__global__ __launch_bounds__(256) void build_qk_norm(
    const bf16* __restrict__ up,   // (BL, NH*DH)  : q_c or k_c
    const bf16* __restrict__ rp,   // q: (BL, NH*DHR); k: (BL, DHR)
    bf16* __restrict__ out,        // (B*NH, L, 192)
    const float* __restrict__ sqk,
    int is_k) {
  const int gid  = blockIdx.x * blockDim.x + threadIdx.x;
  const int lane = gid & 31;
  const int grp  = gid >> 5;            // ((b*L + l)*NH + h)
  const int h  = grp & (NH_ - 1);
  const int bl = grp >> 4;              // b*L + l
  const int l  = bl & (L_ - 1);
  const int b  = bl >> 11;

  float v[6];
#pragma unroll
  for (int t = 0; t < 6; ++t) {
    int d = lane + 32 * t;
    float val;
    if (d < DH_) {
      val = (float)up[(size_t)bl * (NH_ * DH_) + h * DH_ + d];
    } else {
      int j  = d - DH_;                 // 0..63 rope channel
      int jj = j & 31;
      // inv_freq = 10000^(-2*jj/64)
      float inv = __expf(-(float)(2 * jj) * (9.210340371976184f / 64.0f));
      float ang = (float)l * inv;
      float cs = cosf(ang), sn = sinf(ang);
      size_t base = is_k ? (size_t)bl * DHR_
                         : (size_t)bl * (NH_ * DHR_) + (size_t)h * DHR_;
      float xj = (float)rp[base + j];
      float xp = (float)rp[base + (j ^ 32)];
      val = (j < 32) ? (xj * cs - xp * sn) : (xj * cs + xp * sn);
    }
    v[t] = val;
  }
  float ss = 0.f;
#pragma unroll
  for (int t = 0; t < 6; ++t) ss += v[t] * v[t];
#pragma unroll
  for (int o = 16; o >= 1; o >>= 1) ss += __shfl_xor(ss, o, 32);
  float sc = is_k ? 1.0f : sqk[0];
  float rn = sc / fmaxf(sqrtf(ss), 1e-12f);
  bf16* op = out + ((size_t)(b * NH_ + h) * L_ + l) * QKD_;
#pragma unroll
  for (int t = 0; t < 6; ++t) op[lane + 32 * t] = (bf16)(v[t] * rn);
}

// ---------------- flash attention (causal), WMMA for QK^T and PV ----------------
#define KT_  64
#define PSTR 72   // padded LDS stride (keeps 16B alignment, reduces bank conflicts)

__global__ __launch_bounds__(128) void mla_flash_attn(
    const bf16* __restrict__ Q,    // (B*NH, L, 192) pre-scaled by s_qk, unit-norm
    const bf16* __restrict__ Kn,   // (B*NH, L, 192) unit-norm
    const bf16* __restrict__ V,    // (B, L, NH*DH)
    bf16* __restrict__ O) {        // (B, L, NH*DH)
  __shared__ bf16 Vt[DH_][PSTR];     // V tile transposed: [channel][key]
  __shared__ bf16 Ps[4][16][PSTR];   // per-wave P scratch (layout conversion)

  const int tid  = threadIdx.x;
  const int lane = tid & 31;
  const int wave = tid >> 5;          // 4 waves, 16 query rows each
  const int bh = blockIdx.x >> 5;     // L/64 = 32 query tiles per (b,h)
  const int qt = blockIdx.x & 31;
  const int q0 = qt * 64;
  const int b  = bh >> 4;
  const int h  = bh & 15;
  const int hi = lane >> 4;
  const int ln = lane & 15;
  const int qrow = q0 + wave * 16 + ln;   // A-operand row for this lane

  // Q fragments held in registers: K-dim 192 = 6 x 32
  v16bf qf[6];
  {
    const bf16* qb = Q + ((size_t)bh * L_ + qrow) * QKD_;
#pragma unroll
    for (int f = 0; f < 6; ++f) qf[f] = load_frag16(qb + hi * 8 + 32 * f);
  }

  float m_i[8], l_i[8];
  v8f of[8];                         // 8 channel subtiles x (8 rows per lane)
#pragma unroll
  for (int r = 0; r < 8; ++r) { m_i[r] = -1e30f; l_i[r] = 0.f; }
#pragma unroll
  for (int c = 0; c < 8; ++c) of[c] = {};

  const int nkt = qt + 1;            // causal: key tiles up to the diagonal
  for (int kt = 0; kt < nkt; ++kt) {
    const int k0 = kt * KT_;
    __syncthreads();                 // previous Vt consumers done
    // stage V tile transposed into LDS
    {
      const int kk = tid & 63;
      const int c0 = (tid >> 6) * 64;
      const bf16* vb = V + ((size_t)b * L_ + (k0 + kk)) * D_ + h * DH_ + c0;
#pragma unroll
      for (int c8 = 0; c8 < 64; c8 += 8) {
        bf16x8 t = *(const bf16x8*)(vb + c8);
#pragma unroll
        for (int j = 0; j < 8; ++j) Vt[c0 + c8 + j][kk] = t[j];
      }
    }
    __syncthreads();

    // S = Q . K^T : 16x64 per wave, B-operand columns = K rows (contiguous)
    v8f s[4];
#pragma unroll
    for (int ns = 0; ns < 4; ++ns) s[ns] = {};
#pragma unroll
    for (int ns = 0; ns < 4; ++ns) {
      const bf16* kp = Kn + ((size_t)bh * L_ + (k0 + ns * 16 + ln)) * QKD_;
#pragma unroll
      for (int f = 0; f < 6; ++f) {
        v16bf kf = load_frag16(kp + hi * 8 + 32 * f);
        s[ns] = wmma_bf16f32(qf[f], kf, s[ns]);
      }
    }

    // causal mask + online softmax (row = r + 8*hi, col = ns*16 + ln)
#pragma unroll
    for (int r = 0; r < 8; ++r) {
      const int grow = q0 + wave * 16 + r + 8 * hi;
      float mx = -1e30f;
#pragma unroll
      for (int ns = 0; ns < 4; ++ns) {
        const int col = k0 + ns * 16 + ln;
        if (col > grow) s[ns][r] = -1e30f;
        mx = fmaxf(mx, s[ns][r]);
      }
#pragma unroll
      for (int o = 8; o >= 1; o >>= 1) mx = fmaxf(mx, __shfl_xor(mx, o, 16));
      const float newm  = fmaxf(m_i[r], mx);
      const float alpha = __expf(m_i[r] - newm);
      m_i[r] = newm;
      float rs = 0.f;
#pragma unroll
      for (int ns = 0; ns < 4; ++ns) {
        float p = __expf(s[ns][r] - newm);
        s[ns][r] = p;
        rs += p;
      }
#pragma unroll
      for (int o = 8; o >= 1; o >>= 1) rs += __shfl_xor(rs, o, 16);
      l_i[r] = l_i[r] * alpha + rs;
#pragma unroll
      for (int c = 0; c < 8; ++c) of[c][r] *= alpha;
    }

    // P (accumulator layout) -> per-wave LDS -> reload as A-operand
#pragma unroll
    for (int ns = 0; ns < 4; ++ns)
#pragma unroll
      for (int r = 0; r < 8; ++r)
        Ps[wave][r + 8 * hi][ns * 16 + ln] = (bf16)s[ns][r];
    asm volatile("s_wait_dscnt 0x0" ::: "memory");   // wave-private LDS RAW fence

    v16bf pf[2];
#pragma unroll
    for (int f = 0; f < 2; ++f)
      pf[f] = load_frag16(&Ps[wave][ln][hi * 8 + 32 * f]);
#pragma unroll
    for (int c = 0; c < 8; ++c)
#pragma unroll
      for (int f = 0; f < 2; ++f) {
        v16bf vf = load_frag16(&Vt[c * 16 + ln][hi * 8 + 32 * f]);
        of[c] = wmma_bf16f32(pf[f], vf, of[c]);
      }
  }

  // epilogue: divide by softmax denom, store bf16 (B, L, NH*DH)
  float inv[8];
#pragma unroll
  for (int r = 0; r < 8; ++r) inv[r] = 1.0f / fmaxf(l_i[r], 1e-20f);
#pragma unroll
  for (int c = 0; c < 8; ++c)
#pragma unroll
    for (int r = 0; r < 8; ++r) {
      const int row = q0 + wave * 16 + r + 8 * hi;
      const int ch  = c * 16 + ln;
      O[((size_t)b * L_ + row) * D_ + h * DH_ + ch] = (bf16)(of[c][r] * inv[r]);
    }
}

// ---------------- host orchestration ----------------
extern "C" void kernel_launch(void* const* d_in, const int* in_sizes, int n_in,
                              void* d_out, int out_size, void* d_ws, size_t ws_size,
                              hipStream_t stream) {
  (void)in_sizes; (void)n_in; (void)out_size; (void)ws_size;
  const float* x    = (const float*)d_in[0];
  const float* Wdkv = (const float*)d_in[1];
  const float* Wuk  = (const float*)d_in[2];
  const float* Wuv  = (const float*)d_in[3];
  const float* Wdq  = (const float*)d_in[4];
  const float* Wuq  = (const float*)d_in[5];
  const float* Wqr  = (const float*)d_in[6];
  const float* Wkr  = (const float*)d_in[7];
  const float* Wo   = (const float*)d_in[8];
  const float* sqk  = (const float*)d_in[9];

  char* base = (char*)d_ws;
  size_t off = 0;
  auto take = [&](size_t elems) -> bf16* {
    bf16* p = (bf16*)(base + off);
    off += ((elems * sizeof(bf16)) + 255) & ~(size_t)255;
    return p;
  };
  bf16* x_bf   = take((size_t)BL_ * D_);
  bf16* wdkv_b = take((size_t)D_ * DC_);
  bf16* wuk_b  = take((size_t)DC_ * D_);
  bf16* wuv_b  = take((size_t)DC_ * D_);
  bf16* wdq_b  = take((size_t)D_ * DC_);
  bf16* wuq_b  = take((size_t)DC_ * D_);
  bf16* wqr_b  = take((size_t)DC_ * (NH_ * DHR_));
  bf16* wkr_b  = take((size_t)D_ * DHR_);
  bf16* wo_b   = take((size_t)D_ * D_);
  bf16* c_kv   = take((size_t)BL_ * DC_);
  bf16* k_c    = take((size_t)BL_ * D_);
  bf16* v      = take((size_t)BL_ * D_);
  bf16* c_q    = take((size_t)BL_ * DC_);
  bf16* q_c    = take((size_t)BL_ * D_);
  bf16* q_r    = take((size_t)BL_ * (NH_ * DHR_));
  bf16* k_r    = take((size_t)BL_ * DHR_);
  bf16* Qn     = take((size_t)(B_ * NH_) * L_ * QKD_);
  bf16* Kn     = take((size_t)(B_ * NH_) * L_ * QKD_);
  bf16* att_o  = take((size_t)BL_ * D_);

  auto cast = [&](const float* in, bf16* out, size_t n) {
    int blocks = (int)((n + 1023) / 1024);
    if (blocks > 4096) blocks = 4096;
    cast_f32_to_bf16<<<blocks, 256, 0, stream>>>(in, out, (int)n);
  };
  cast(x,    x_bf,   (size_t)BL_ * D_);
  cast(Wdkv, wdkv_b, (size_t)D_ * DC_);
  cast(Wuk,  wuk_b,  (size_t)DC_ * D_);
  cast(Wuv,  wuv_b,  (size_t)DC_ * D_);
  cast(Wdq,  wdq_b,  (size_t)D_ * DC_);
  cast(Wuq,  wuq_b,  (size_t)DC_ * D_);
  cast(Wqr,  wqr_b,  (size_t)DC_ * (NH_ * DHR_));
  cast(Wkr,  wkr_b,  (size_t)D_ * DHR_);
  cast(Wo,   wo_b,   (size_t)D_ * D_);

  const size_t gemm_lds = 2 * (size_t)GBM * GBK * sizeof(bf16);  // 16 KB
  auto gemm_b = [&](const bf16* A, const bf16* Bw, bf16* C, int M, int N, int K) {
    dim3 g((N + GBN - 1) / GBN, M / GBM);
    gemm_bf16_wmma<bf16><<<g, 256, gemm_lds, stream>>>(A, Bw, C, M, N, K);
  };
  gemm_b(x_bf, wdkv_b, c_kv, BL_, DC_, D_);            // c_kv = x @ W_DKV
  gemm_b(c_kv, wuk_b,  k_c,  BL_, D_,  DC_);           // k_c
  gemm_b(c_kv, wuv_b,  v,    BL_, D_,  DC_);           // v
  gemm_b(x_bf, wdq_b,  c_q,  BL_, DC_, D_);            // c_q
  gemm_b(c_q,  wuq_b,  q_c,  BL_, D_,  DC_);           // q_c
  gemm_b(c_q,  wqr_b,  q_r,  BL_, NH_ * DHR_, DC_);    // q_rope pre
  gemm_b(x_bf, wkr_b,  k_r,  BL_, DHR_, D_);           // k_rope pre (N=64, guarded)

  const int qk_blocks = (B_ * L_ * NH_ * 32) / 256;    // one wave per (b,l,h)
  build_qk_norm<<<qk_blocks, 256, 0, stream>>>(q_c, q_r, Qn, sqk, 0);
  build_qk_norm<<<qk_blocks, 256, 0, stream>>>(k_c, k_r, Kn, sqk, 1);

  mla_flash_attn<<<B_ * NH_ * (L_ / 64), 128, 0, stream>>>(Qn, Kn, v, att_o);

  dim3 g2((D_ + GBN - 1) / GBN, BL_ / GBM);
  gemm_bf16_wmma<float><<<g2, 256, gemm_lds, stream>>>(att_o, wo_b, (float*)d_out,
                                                       BL_, D_, D_);
}
